// DRMM_14173392076891
// MI455X (gfx1250) — matrix operand order repaired and checked by reading, not compiled, not run on hardware
//
#include <hip/hip_runtime.h>

#define B_      32
#define D_      8
#define QL_     16
#define DL_     512
#define E_      300
#define KPAD    320        // E padded to multiple of 32 (WMMA K); zero-filled
#define KSTRIDE 328        // LDS row stride in halves: 164 dwords -> conflict-free banks
#define NTILE   4          // doc tiles per iteration (one per wave)
#define ITERS   8          // 512 / (NTILE*16)

typedef _Float16 half8  __attribute__((ext_vector_type(8)));
typedef _Float16 half16 __attribute__((ext_vector_type(16)));
typedef float    float8 __attribute__((ext_vector_type(8)));

// ---------------- Kernel 1: gating network + softmax over query terms ----------------
__global__ __launch_bounds__(32) void drmm_gate_kernel(
    const int* __restrict__ bq, const float* __restrict__ emb,
    const float* __restrict__ w_g, const float* __restrict__ b_g,
    float* __restrict__ tw)
{
  const int b = blockIdx.x;
  const int q = threadIdx.x;
  __shared__ float g[QL_];
  __shared__ float red[2];
  if (q < QL_) {
    const int tok = bq[b * QL_ + q];
    const float* row = emb + (long)tok * E_;
    float acc = 0.f;
    for (int e = 0; e < E_; ++e) acc += row[e] * w_g[e];
    g[q] = acc + b_g[0];
  }
  __syncthreads();
  if (q == 0) {
    float mx = g[0];
    for (int i = 1; i < QL_; ++i) mx = fmaxf(mx, g[i]);
    float s = 0.f;
    for (int i = 0; i < QL_; ++i) s += __expf(g[i] - mx);
    red[0] = mx; red[1] = s;
  }
  __syncthreads();
  if (q < QL_) tw[b * QL_ + q] = __expf(g[q] - red[0]) / red[1];
}

// ---------------- Kernel 2: WMMA cosine matrix + histogram + score ----------------
__global__ __launch_bounds__(128) void drmm_main_kernel(
    const int* __restrict__ bq, const int* __restrict__ bd,
    const float* __restrict__ emb,
    const float* __restrict__ w1, const float* __restrict__ b1,
    const float* __restrict__ w2, const float* __restrict__ b2,
    const float* __restrict__ w_o, const float* __restrict__ b_o,
    const float* __restrict__ tw,
    float* __restrict__ out)
{
  __shared__ __align__(16) _Float16 Qs[QL_][KSTRIDE];          // ~10.3 KB
  __shared__ __align__(16) _Float16 Ds[NTILE][16][KSTRIDE];    // ~41 KB
  __shared__ float qn[QL_];
  __shared__ float dn[NTILE][16];
  __shared__ int   hist[QL_][5];
  __shared__ float red[QL_];

  const int tid = threadIdx.x;
  const int bn  = blockIdx.x;        // 0..255 == b*D + d (docs flattened)
  const int b   = bn >> 3;           // D_ == 8

  if (tid < QL_) qn[tid] = 0.f;
  if (tid < QL_ * 5) ((int*)hist)[tid] = 0;
  __syncthreads();

  // ---- stage Q tile: 16 tokens x 320 halves (row stride 328); f32 norms ----
  {
    const int q   = tid >> 3;        // 0..15
    const int sub = tid & 7;         // each covers 10 float4 slots (of 80)
    const int tok = bq[b * QL_ + q];
    const float4* row = (const float4*)(emb + (long)tok * E_);  // 1200 % 16 == 0
    float ss = 0.f;
    for (int i = 0; i < 10; ++i) {
      const int f4 = sub * 10 + i;   // 0..79; valid < 75 (=300 floats)
      float4 v = make_float4(0.f, 0.f, 0.f, 0.f);
      if (f4 < 75) v = row[f4];
      ss += v.x*v.x + v.y*v.y + v.z*v.z + v.w*v.w;
      _Float16* dst = &Qs[q][f4 * 4];
      dst[0] = (_Float16)v.x; dst[1] = (_Float16)v.y;
      dst[2] = (_Float16)v.z; dst[3] = (_Float16)v.w;
    }
    atomicAdd(&qn[q], ss);
  }
  __syncthreads();
  if (tid < QL_) qn[tid] = sqrtf(qn[tid]);

  // wave32 WMMA operand coordinates
  const int l  = tid & 31;
  const int w  = tid >> 5;           // wave -> doc tile
  const int ma = l & 15;             // A: row M per lane
  const int kb = (l >> 4) * 8;       // A: K sub-offset (0 / 8)
  const int nn = l & 15;             // B/C: column N per lane
  const int kh = (l >> 4) * 8;       // B: K sub-offset (0 / 8)
  const int mc = (l >> 4) * 8;       // C: row base (0 / 8)

  // ---- hoist the 10 A-fragments (Q is invariant across all doc iterations) ----
  half16 afr[KPAD / 32];
  #pragma unroll
  for (int s = 0; s < KPAD / 32; ++s) {
    half8 a0 = *(const half8*)&Qs[ma][s * 32 + kb];        // K = kb..kb+7
    half8 a1 = *(const half8*)&Qs[ma][s * 32 + kb + 16];   // K = kb+16..kb+23
    afr[s] = __builtin_shufflevector(a0, a1,
        0, 1, 2, 3, 4, 5, 6, 7, 8, 9, 10, 11, 12, 13, 14, 15);
  }

  const int* docRow = bd + (long)bn * DL_;

  for (int it = 0; it < ITERS; ++it) {
    __syncthreads();                                   // prev compute done
    if (tid < NTILE * 16) ((float*)dn)[tid] = 0.f;
    __syncthreads();

    // ---- gather 64 doc tokens (4 tiles x 16) into f16 LDS ----
    {
      const int tt  = tid >> 1;      // 0..63
      const int sub = tid & 1;       // each covers 40 float4 slots
      const int tok = docRow[it * 64 + tt];
      const int tl  = tt >> 4;       // tile
      const int tn  = tt & 15;       // token within tile
      const float4* row = (const float4*)(emb + (long)tok * E_);
      float ss = 0.f;
      for (int i = 0; i < 40; ++i) {
        const int f4 = sub * 40 + i;
        float4 v = make_float4(0.f, 0.f, 0.f, 0.f);
        if (f4 < 75) v = row[f4];
        ss += v.x*v.x + v.y*v.y + v.z*v.z + v.w*v.w;
        _Float16* dst = &Ds[tl][tn][f4 * 4];
        dst[0] = (_Float16)v.x; dst[1] = (_Float16)v.y;
        dst[2] = (_Float16)v.z; dst[3] = (_Float16)v.w;
      }
      atomicAdd(&dn[tl][tn], ss);
      if (it + 1 < ITERS && sub == 0) {                // warm caches for next tiles
        const int ntok = docRow[(it + 1) * 64 + tt];
        __builtin_prefetch(emb + (long)ntok * E_, 0, 1);
      }
    }
    __syncthreads();
    if (tid < NTILE * 16) ((float*)dn)[tid] = sqrtf(((float*)dn)[tid]);
    __syncthreads();

    // ---- 16x16 dot tile per wave over K=320 via v_wmma_f32_16x16x32_f16 ----
    float8 acc = {0.f, 0.f, 0.f, 0.f, 0.f, 0.f, 0.f, 0.f};
    #pragma unroll
    for (int s = 0; s < KPAD / 32; ++s) {
      const int kk = s * 32;
      half8 b0 = *(const half8*)&Ds[w][nn][kk + kh];      // K = kh..kh+7
      half8 b1 = *(const half8*)&Ds[w][nn][kk + kh + 16]; // K = kh+16..kh+23
      half16 bfrag = __builtin_shufflevector(b0, b1,
          0, 1, 2, 3, 4, 5, 6, 7, 8, 9, 10, 11, 12, 13, 14, 15);
      acc = __builtin_amdgcn_wmma_f32_16x16x32_f16(
          false, afr[s], false, bfrag, (short)0, acc, false, false);
    }

    // ---- branchless, division-free histogram: cos ? t  <=>  dot ? t*denom ----
    const float dnv = dn[w][nn];
    #pragma unroll
    for (int r = 0; r < 8; ++r) {
      const int m = mc + r;
      const float denom = fmaxf(qn[m] * dnv, 1e-8f);   // > 0 always
      const float hd    = 0.5f * denom;
      const float a     = acc[r];
      // ordered-edge predicate sum: [-1,-.5) [-0.5,0) [0,.5) [.5,1) [1,1]
      const int bin = (int)(a >= -hd) + (int)(a >= 0.0f)
                    + (int)(a >=  hd) + (int)(a >= denom);
      if (a >= -denom && a <= denom)                   // numpy: drop out-of-range
        atomicAdd(&hist[m][bin], 1);
    }
  }
  __syncthreads();

  // ---- FFNN + gating + output score ----
  if (tid < QL_) {
    const int m = tid;
    float f = 0.f;
    #pragma unroll
    for (int kbin = 0; kbin < 5; ++kbin)
      f += (float)hist[m][kbin] * w1[kbin];
    f = (f + b1[0]) * w2[0] + b2[0];
    red[m] = f * tw[b * QL_ + m];
  }
  __syncthreads();
  if (tid == 0) {
    float s = 0.f;
    for (int i = 0; i < QL_; ++i) s += red[i];
    out[bn] = s * w_o[0] + b_o[0];
  }
}

// ---------------- host launcher ----------------
extern "C" void kernel_launch(void* const* d_in, const int* in_sizes, int n_in,
                              void* d_out, int out_size, void* d_ws, size_t ws_size,
                              hipStream_t stream) {
  const int*   bq  = (const int*)d_in[0];    // batch_queries [32,16]
  // d_in[1] query_len: unused by reference
  const int*   bd  = (const int*)d_in[2];    // batch_docs [32,8,512]
  // d_in[3] doc_len: unused by reference
  const float* emb = (const float*)d_in[4];  // [50000,300]
  const float* w_g = (const float*)d_in[5];
  const float* b_g = (const float*)d_in[6];
  const float* w1  = (const float*)d_in[7];
  const float* b1  = (const float*)d_in[8];
  const float* w2  = (const float*)d_in[9];
  const float* b2  = (const float*)d_in[10];
  const float* w_o = (const float*)d_in[11];
  const float* b_o = (const float*)d_in[12];
  float* out = (float*)d_out;                // [32,8]
  float* tw  = (float*)d_ws;                 // 512 floats term weights

  drmm_gate_kernel<<<B_, 32, 0, stream>>>(bq, emb, w_g, b_g, tw);
  drmm_main_kernel<<<B_ * D_, 128, 0, stream>>>(bq, bd, emb, w1, b1, w2, b2,
                                                w_o, b_o, tw, out);
}
